// MultiHeadSelfAttention_5609227289113
// MI455X (gfx1250) — matrix operand order repaired and checked by reading, not compiled, not run on hardware
//
#include <hip/hip_runtime.h>

typedef __bf16 bhalf;
typedef __attribute__((ext_vector_type(16))) __bf16 v16bf;
typedef __attribute__((ext_vector_type(8)))  __bf16 v8bf;
typedef __attribute__((ext_vector_type(8)))  float   v8f;

constexpr int Bc  = 2;
constexpr int Tc  = 2048;
constexpr int Dc  = 2048;
constexpr int Hc  = 16;
constexpr int HDc = 128;
constexpr int Mc  = Bc * Tc;        // 4096 rows of x
constexpr int N3  = Hc * 3 * HDc;   // 6144 qkv output cols
constexpr int PSTR = 40;            // padded LDS row stride (bf16 elems)
constexpr size_t PAD = 512;         // tail pad per ws buffer (ping-pong over-read)

// ---------------- helpers ----------------

__device__ __forceinline__ v16bf ld16(const bhalf* p0, const bhalf* p1) {
  v8bf a = *(const v8bf*)p0;
  v8bf b = *(const v8bf*)p1;
  return __builtin_shufflevector(a, b, 0,1,2,3,4,5,6,7,8,9,10,11,12,13,14,15);
}
// A operand (16x32 bf16): lane = kg*16+m holds row m; VGPR j<4: K=kg*8+2j, j>=4: K=16+kg*8+...
__device__ __forceinline__ v16bf ldA(const bhalf* row, int k0, int kg) {
  return ld16(row + k0 + kg * 8, row + k0 + kg * 8 + 16);
}
// B operand (32x16 bf16): lane = kg*16+n holds col n; VGPR j<4: K=kg*8+2j, j>=4: K=16+kg*8+...
__device__ __forceinline__ v16bf ldB(const bhalf* row, int k0, int kg) {
  return ld16(row + k0 + kg * 8, row + k0 + kg * 8 + 16);
}
__device__ __forceinline__ v8f wmma_bf16(v16bf a, v16bf b, v8f c) {
  return __builtin_amdgcn_wmma_f32_16x16x32_bf16(false, a, false, b, (short)0, c, false, false);
}

// ---------------- conversion kernels ----------------

// vectorized: 8 elems/thread, one b128 store
__global__ void cvt_bf16_kernel(const float* __restrict__ src, bhalf* __restrict__ dst, int n8) {
  int i = blockIdx.x * blockDim.x + threadIdx.x;
  if (i < n8) {
    const float4* s = (const float4*)src + 2 * (size_t)i;
    float4 f0 = s[0], f1 = s[1];
    v8bf o;
    o[0] = (bhalf)f0.x; o[1] = (bhalf)f0.y; o[2] = (bhalf)f0.z; o[3] = (bhalf)f0.w;
    o[4] = (bhalf)f1.x; o[5] = (bhalf)f1.y; o[6] = (bhalf)f1.z; o[7] = (bhalf)f1.w;
    ((v8bf*)dst)[i] = o;
  }
}

// w: [rows][cols] f32  ->  wt: [cols][rows] bf16   (rows, cols multiples of 32)
__global__ void transpose_cvt_kernel(const float* __restrict__ w, bhalf* __restrict__ wt,
                                     int rows, int cols) {
  __shared__ float tile[32][33];
  int r0 = blockIdx.y * 32, c0 = blockIdx.x * 32;
  int tx = threadIdx.x, ty = threadIdx.y;
  for (int i = ty; i < 32; i += 8)
    tile[i][tx] = w[(size_t)(r0 + i) * cols + (c0 + tx)];
  __syncthreads();
  for (int i = ty; i < 32; i += 8)
    wt[(size_t)(c0 + i) * rows + (r0 + tx)] = (bhalf)tile[tx][i];
}

// ---------------- QKV GEMM + fused RoPE ----------------
// xb:[Mc][Dc] bf16, wt:[N3][Dc] bf16 (transposed w_qkv)
// Qb,Kb: [B][H][T][HD] bf16 (RoPE applied), Vt: [B][H][HD][T] bf16
__global__ void __launch_bounds__(128) qkv_rope_kernel(const bhalf* __restrict__ xb,
                                                       const bhalf* __restrict__ wt,
                                                       bhalf* __restrict__ Qb,
                                                       bhalf* __restrict__ Kb,
                                                       bhalf* __restrict__ Vt) {
  const int lane = threadIdx.x & 31;
  const int wv   = threadIdx.x >> 5;
  const int ln16 = lane & 15;
  const int kg   = lane >> 4;
  const int Mbase = blockIdx.y * 64 + (wv & 1) * 32;
  const int Nbase = blockIdx.x * 128 + (wv >> 1) * 64;

  v8f C[2][4];
  const v8f vz = {0.f,0.f,0.f,0.f,0.f,0.f,0.f,0.f};
  for (int i = 0; i < 2; ++i)
    for (int j = 0; j < 4; ++j) C[i][j] = vz;

  const bhalf* a0 = xb + (size_t)(Mbase + ln16) * Dc;
  const bhalf* a1 = a0 + (size_t)16 * Dc;
  const bhalf* brow[4];
  for (int j = 0; j < 4; ++j) brow[j] = wt + (size_t)(Nbase + j * 16 + ln16) * Dc;

  // ping-pong pipelined K loop (x2 unroll -> no buffer-rotate copies)
  v16bf A0a = ldA(a0, 0, kg);
  v16bf A1a = ldA(a1, 0, kg);
  v16bf Ba[4];
#pragma unroll
  for (int j = 0; j < 4; ++j) Ba[j] = ldB(brow[j], 0, kg);

  for (int k0 = 0; k0 < Dc; k0 += 64) {
    v16bf A0b = ldA(a0, k0 + 32, kg);
    v16bf A1b = ldA(a1, k0 + 32, kg);
    v16bf Bb[4];
#pragma unroll
    for (int j = 0; j < 4; ++j) Bb[j] = ldB(brow[j], k0 + 32, kg);
    __builtin_prefetch(a0 + k0 + 128, 0, 1);        // CDNA5 global_prefetch_b8
    __builtin_prefetch(brow[0] + k0 + 128, 0, 1);
#pragma unroll
    for (int j = 0; j < 4; ++j) {
      C[0][j] = wmma_bf16(A0a, Ba[j], C[0][j]);
      C[1][j] = wmma_bf16(A1a, Ba[j], C[1][j]);
    }
    // last iteration over-reads one K-chunk past Dc (ws buffers are padded)
    A0a = ldA(a0, k0 + 64, kg);
    A1a = ldA(a1, k0 + 64, kg);
#pragma unroll
    for (int j = 0; j < 4; ++j) Ba[j] = ldB(brow[j], k0 + 64, kg);
#pragma unroll
    for (int j = 0; j < 4; ++j) {
      C[0][j] = wmma_bf16(A0b, Bb[j], C[0][j]);
      C[1][j] = wmma_bf16(A1b, Bb[j], C[1][j]);
    }
  }

  const float LN_THETA_OVER_HD = 0.07195571856f; // ln(10000)/128
#pragma unroll
  for (int mt = 0; mt < 2; ++mt) {
#pragma unroll
    for (int nt = 0; nt < 4; ++nt) {
      const int n0 = Nbase + nt * 16;
      const int h  = n0 / 384;           // uniform across wave (tile never straddles)
      const int fb = n0 % 384;
      const int sec = fb >> 7;           // 0=q, 1=k, 2=v
      const int d = (fb & 127) + ln16;   // 0..127 within head
      v8f c = C[mt][nt];
      if (sec < 2) {
        const int i2 = d & ~1;
        const float invf = __expf(-(float)i2 * LN_THETA_OVER_HD);
        bhalf* dst = (sec == 0) ? Qb : Kb;
#pragma unroll
        for (int j = 0; j < 8; ++j) {
          int m = j + 8 * kg;
          int tg = Mbase + mt * 16 + m;
          int b = tg >> 11, t = tg & (Tc - 1);
          float ang = (float)t * invf;
          float sn, cs;
          __sincosf(ang, &sn, &cs);
          float v  = c[j];
          float pv = __shfl_xor(v, 1, 32);   // RoPE interleaved partner
          float o  = (d & 1) ? (pv * sn + v * cs) : (v * cs - pv * sn);
          dst[((size_t)(b * Hc + h) * Tc + t) * HDc + d] = (bhalf)o;
        }
      } else {
#pragma unroll
        for (int j = 0; j < 8; ++j) {
          int m = j + 8 * kg;
          int tg = Mbase + mt * 16 + m;
          int b = tg >> 11, t = tg & (Tc - 1);
          Vt[((size_t)(b * Hc + h) * HDc + d) * Tc + t] = (bhalf)c[j];
        }
      }
    }
  }
}

// ---------------- flash attention ----------------
// Qb,Kb:[B][H][T][HD], Vt:[B][H][HD][T], Yb:[B][T][D] (head-concat layout)
__global__ void __launch_bounds__(128) attn_kernel(const bhalf* __restrict__ Qb,
                                                   const bhalf* __restrict__ Kb,
                                                   const bhalf* __restrict__ Vt,
                                                   bhalf* __restrict__ Yb) {
  __shared__ __align__(16) bhalf Plds[4][16 * PSTR];
  const int lane = threadIdx.x & 31;
  const int wv   = threadIdx.x >> 5;
  const int ln16 = lane & 15;
  const int kg   = lane >> 4;
  const int qt    = blockIdx.x * 4 + wv;
  const int qbase = qt * 16;
  const int h = blockIdx.y, b = blockIdx.z;
  const size_t bh = (size_t)(b * Hc + h);

  const bhalf* qrow = Qb + (bh * Tc + qbase + ln16) * HDc;
  v16bf aQ[4];
#pragma unroll
  for (int kc = 0; kc < 4; ++kc) aQ[kc] = ldA(qrow, kc * 32, kg);

  const v8f vz = {0.f,0.f,0.f,0.f,0.f,0.f,0.f,0.f};
  v8f O[8];
  float mst[8], lst[8];
#pragma unroll
  for (int j = 0; j < 8; ++j) { O[j] = vz; mst[j] = -1e30f; lst[j] = 0.f; }

  const float sm_scale = 0.08838834764831845f; // 1/sqrt(128)
  const int nkb = (qbase + 16 + 31) >> 5;      // causal key-block count

  for (int kb = 0; kb < nkb; ++kb) {
    const int key0 = kb << 5;
    v8f S0 = vz, S1 = vz;
    const bhalf* kr0 = Kb + (bh * Tc + key0 + ln16) * HDc;
    const bhalf* kr1 = kr0 + (size_t)16 * HDc;
#pragma unroll
    for (int kc = 0; kc < 4; ++kc) {
      v16bf b0 = ldB(kr0, kc * 32, kg);
      v16bf b1 = ldB(kr1, kc * 32, kg);
      S0 = wmma_bf16(aQ[kc], b0, S0);
      S1 = wmma_bf16(aQ[kc], b1, S1);
    }

    // Issue V loads now: independent of softmax, latency hides under the VALU below.
    v16bf bV[8];
    const bhalf* vbase = Vt + (bh * HDc + ln16) * Tc + key0 + kg * 8;
#pragma unroll
    for (int dt = 0; dt < 8; ++dt) {
      const bhalf* vp = vbase + (size_t)dt * 16 * Tc;
      bV[dt] = ld16(vp, vp + 16);
    }
    if (kb + 1 < nkb) {                      // prefetch next K block rows
      __builtin_prefetch(kr0 + (size_t)32 * HDc, 0, 1);
      __builtin_prefetch(kr1 + (size_t)32 * HDc, 0, 1);
    }

    bhalf* pl = Plds[wv];
#pragma unroll
    for (int j = 0; j < 8; ++j) {
      const int m  = j + 8 * kg;
      const int qi = qbase + m;
      float s0 = S0[j] * sm_scale;
      float s1 = S1[j] * sm_scale;
      if (key0 + ln16 > qi)      s0 = -1e30f;   // causal mask
      if (key0 + 16 + ln16 > qi) s1 = -1e30f;
      float rmax = fmaxf(s0, s1);
#pragma unroll
      for (int msk = 1; msk < 16; msk <<= 1)
        rmax = fmaxf(rmax, __shfl_xor(rmax, msk, 32));
      const float mnew = fmaxf(mst[j], rmax);
      const float corr = __expf(mst[j] - mnew);
      mst[j] = mnew;
      const float e0 = __expf(s0 - mnew);
      const float e1 = __expf(s1 - mnew);
      float rs = e0 + e1;
#pragma unroll
      for (int msk = 1; msk < 16; msk <<= 1) rs += __shfl_xor(rs, msk, 32);
      lst[j] = lst[j] * corr + rs;
#pragma unroll
      for (int dt = 0; dt < 8; ++dt) O[dt][j] *= corr;
      pl[m * PSTR + ln16]      = (bhalf)e0;
      pl[m * PSTR + 16 + ln16] = (bhalf)e1;
    }
    asm volatile("s_wait_dscnt 0" ::: "memory");   // same-wave LDS RAW turnaround

    v16bf aP = ld16(pl + ln16 * PSTR + kg * 8, pl + ln16 * PSTR + kg * 8 + 16);
#pragma unroll
    for (int dt = 0; dt < 8; ++dt)
      O[dt] = wmma_bf16(aP, bV[dt], O[dt]);
  }

#pragma unroll
  for (int j = 0; j < 8; ++j) lst[j] = 1.0f / lst[j];
#pragma unroll
  for (int dt = 0; dt < 8; ++dt) {
#pragma unroll
    for (int j = 0; j < 8; ++j) {
      const int qi = qbase + j + 8 * kg;
      const float o = O[dt][j] * lst[j];
      Yb[((size_t)b * Tc + qi) * Dc + h * HDc + dt * 16 + ln16] = (bhalf)o;
    }
  }
}

// ---------------- output projection GEMM ----------------
// Yb:[Mc][Dc] bf16, wt:[Dc][Dc] bf16 (transposed w_out), out:[Mc][Dc] f32
__global__ void __launch_bounds__(128) out_gemm_kernel(const bhalf* __restrict__ Yb,
                                                       const bhalf* __restrict__ wt,
                                                       float* __restrict__ out) {
  const int lane = threadIdx.x & 31;
  const int wv   = threadIdx.x >> 5;
  const int ln16 = lane & 15;
  const int kg   = lane >> 4;
  const int Mbase = blockIdx.y * 64 + (wv & 1) * 32;
  const int Nbase = blockIdx.x * 128 + (wv >> 1) * 64;

  v8f C[2][4];
  const v8f vz = {0.f,0.f,0.f,0.f,0.f,0.f,0.f,0.f};
  for (int i = 0; i < 2; ++i)
    for (int j = 0; j < 4; ++j) C[i][j] = vz;

  const bhalf* a0 = Yb + (size_t)(Mbase + ln16) * Dc;
  const bhalf* a1 = a0 + (size_t)16 * Dc;
  const bhalf* brow[4];
  for (int j = 0; j < 4; ++j) brow[j] = wt + (size_t)(Nbase + j * 16 + ln16) * Dc;

  v16bf A0a = ldA(a0, 0, kg);
  v16bf A1a = ldA(a1, 0, kg);
  v16bf Ba[4];
#pragma unroll
  for (int j = 0; j < 4; ++j) Ba[j] = ldB(brow[j], 0, kg);

  for (int k0 = 0; k0 < Dc; k0 += 64) {
    v16bf A0b = ldA(a0, k0 + 32, kg);
    v16bf A1b = ldA(a1, k0 + 32, kg);
    v16bf Bb[4];
#pragma unroll
    for (int j = 0; j < 4; ++j) Bb[j] = ldB(brow[j], k0 + 32, kg);
    __builtin_prefetch(a0 + k0 + 128, 0, 1);
    __builtin_prefetch(brow[0] + k0 + 128, 0, 1);
#pragma unroll
    for (int j = 0; j < 4; ++j) {
      C[0][j] = wmma_bf16(A0a, Ba[j], C[0][j]);
      C[1][j] = wmma_bf16(A1a, Ba[j], C[1][j]);
    }
    A0a = ldA(a0, k0 + 64, kg);
    A1a = ldA(a1, k0 + 64, kg);
#pragma unroll
    for (int j = 0; j < 4; ++j) Ba[j] = ldB(brow[j], k0 + 64, kg);
#pragma unroll
    for (int j = 0; j < 4; ++j) {
      C[0][j] = wmma_bf16(A0b, Bb[j], C[0][j]);
      C[1][j] = wmma_bf16(A1b, Bb[j], C[1][j]);
    }
  }

#pragma unroll
  for (int mt = 0; mt < 2; ++mt) {
#pragma unroll
    for (int nt = 0; nt < 4; ++nt) {
      const int n0 = Nbase + nt * 16;
#pragma unroll
      for (int j = 0; j < 8; ++j) {
        const int m = j + 8 * kg;
        out[(size_t)(Mbase + mt * 16 + m) * Dc + n0 + ln16] = C[mt][nt][j];
      }
    }
  }
}

// ---------------- launch ----------------

extern "C" void kernel_launch(void* const* d_in, const int* in_sizes, int n_in,
                              void* d_out, int out_size, void* d_ws, size_t ws_size,
                              hipStream_t stream) {
  (void)in_sizes; (void)n_in; (void)out_size; (void)ws_size;
  const float* x     = (const float*)d_in[0];
  const float* w_qkv = (const float*)d_in[1];
  const float* w_out = (const float*)d_in[2];
  float* out = (float*)d_out;

  char* ws = (char*)d_ws;
  size_t off = 0;
  bhalf* xb     = (bhalf*)(ws + off); off += (size_t)Mc * Dc * 2 + PAD;   // 16 MiB
  bhalf* wqkvT  = (bhalf*)(ws + off); off += (size_t)N3 * Dc * 2 + PAD;   // 24 MiB
  bhalf* woutT  = (bhalf*)(ws + off); off += (size_t)Dc * Dc * 2 + PAD;   //  8 MiB
  bhalf* Qb     = (bhalf*)(ws + off); off += (size_t)Mc * Dc * 2 + PAD;   // 16 MiB
  bhalf* Kb     = (bhalf*)(ws + off); off += (size_t)Mc * Dc * 2 + PAD;   // 16 MiB
  bhalf* Vt     = (bhalf*)(ws + off); off += (size_t)Mc * Dc * 2 + PAD;   // 16 MiB
  bhalf* Yb     = (bhalf*)(ws + off); off += (size_t)Mc * Dc * 2 + PAD;   // 16 MiB

  const int n8 = (Mc * Dc) / 8;
  cvt_bf16_kernel<<<dim3(n8 / 256), dim3(256), 0, stream>>>(x, xb, n8);
  transpose_cvt_kernel<<<dim3(N3 / 32, Dc / 32), dim3(32, 8), 0, stream>>>(w_qkv, wqkvT, Dc, N3);
  transpose_cvt_kernel<<<dim3(Dc / 32, Dc / 32), dim3(32, 8), 0, stream>>>(w_out, woutT, Dc, Dc);
  qkv_rope_kernel<<<dim3(N3 / 128, Mc / 64), dim3(128), 0, stream>>>(xb, wqkvT, Qb, Kb, Vt);
  attn_kernel<<<dim3(Tc / 64, Hc, Bc), dim3(128), 0, stream>>>(Qb, Kb, Vt, Yb);
  out_gemm_kernel<<<dim3(Dc / 128, Mc / 64), dim3(128), 0, stream>>>(Yb, woutT, out);
}
